// DGT_21371757265626
// MI455X (gfx1250) — matrix operand order repaired and checked by reading, not compiled, not run on hardware
//
#include <hip/hip_runtime.h>

typedef __attribute__((ext_vector_type(16))) __bf16 v16bf;
typedef __attribute__((ext_vector_type(8)))  __bf16 v8bf;
typedef __attribute__((ext_vector_type(8)))  float  v8f;

#define DGT_BATCH   65536
#define DGT_INDIM   256
#define DGT_NODES   1023
#define DGT_NPAD    1024
#define DGT_LEAVES  1024
#define DGT_OUT     32

// ---------------- workspace layout (bytes) ----------------
// x_bf16   : 65536*256*2  = 33554432
// Wp_bf16  : 1024*256*2   = 524288   (row 1023 zero-padded)
// Wor_bf16 : 32*1024*2    = 65536
// colSum/colSq/sArr/tArr : 4 * 4096
// predZ    : 65536*1024*2 = 134217728   (bf16 -> fits in 192MB L2 with x_bf16)
static constexpr size_t OFF_XB   = 0;
static constexpr size_t OFF_WPB  = 33554432;
static constexpr size_t OFF_WORB = OFF_WPB + 524288;
static constexpr size_t OFF_SUM  = OFF_WORB + 65536;
static constexpr size_t OFF_SQ   = OFF_SUM + 4096;
static constexpr size_t OFF_S    = OFF_SQ + 4096;
static constexpr size_t OFF_T    = OFF_S + 4096;
static constexpr size_t OFF_PZ   = OFF_T + 4096;

// ---------------- prep kernels ----------------
__global__ __launch_bounds__(256) void dgt_cvt_bf16(const float* __restrict__ src,
                                                    __bf16* __restrict__ dst,
                                                    int nsrc, int ntot) {
  for (int i = blockIdx.x * 256 + threadIdx.x; i < ntot; i += gridDim.x * 256)
    dst[i] = (i < nsrc) ? (__bf16)src[i] : (__bf16)0.0f;
}

__global__ __launch_bounds__(256) void dgt_zero_f32(float* __restrict__ p, int n) {
  int i = blockIdx.x * 256 + threadIdx.x;
  if (i < n) p[i] = 0.0f;
}

// ---------------- stage 1: pred GEMM (bf16 WMMA) + BN stats ----------------
// grid = (BATCH/128, NPAD/64); 256 threads = 8 waves.
// Wave w computes a 16x64 output strip (4 WMMA tiles) -> 4x A-fragment reuse.
__global__ __launch_bounds__(256) void dgt_pred_gemm(
    const __bf16* __restrict__ Xb, const __bf16* __restrict__ Wb,
    const float* __restrict__ bp, __bf16* __restrict__ predZ,
    float* __restrict__ colSum, float* __restrict__ colSq)
{
  __shared__ float sSum[64];
  __shared__ float sSq[64];
  const int tid  = threadIdx.x;
  const int w    = tid >> 5;
  const int lane = tid & 31;
  const int m    = lane & 15;   // row-in-tile (A) / col-in-tile (B)
  const int hi   = lane >> 4;   // half-wave selector
  const int rowBase = blockIdx.x * 128 + w * 16;
  const int nBase   = blockIdx.y * 64;

  if (tid < 64) { sSum[tid] = 0.0f; sSq[tid] = 0.0f; }
  __syncthreads();

  const __bf16* arow = Xb + (size_t)(rowBase + m) * DGT_INDIM; // A: row m of x tile
  const __bf16* brow = Wb + (size_t)(nBase + m) * DGT_INDIM;   // B tile 0: col m = Wp row

  v8f c0 = {}, c1 = {}, c2 = {}, c3 = {};
  #pragma unroll
  for (int kk = 0; kk < DGT_INDIM; kk += 32) {
    __builtin_prefetch(arow + kk + 64, 0, 1);
    v8bf a0 = *(const v8bf*)(arow + kk + hi * 16);
    v8bf a1 = *(const v8bf*)(arow + kk + hi * 16 + 8);
    v16bf a;
    #pragma unroll
    for (int i = 0; i < 8; ++i) { a[i] = a0[i]; a[8 + i] = a1[i]; }
    v16bf b0 = *(const v16bf*)(brow + kk + hi * 16);
    v16bf b1 = *(const v16bf*)(brow + 16 * DGT_INDIM + kk + hi * 16);
    v16bf b2 = *(const v16bf*)(brow + 32 * DGT_INDIM + kk + hi * 16);
    v16bf b3 = *(const v16bf*)(brow + 48 * DGT_INDIM + kk + hi * 16);
    c0 = __builtin_amdgcn_wmma_f32_16x16x32_bf16(false, a, false, b0, (short)0, c0, false, false);
    c1 = __builtin_amdgcn_wmma_f32_16x16x32_bf16(false, a, false, b1, (short)0, c1, false, false);
    c2 = __builtin_amdgcn_wmma_f32_16x16x32_bf16(false, a, false, b2, (short)0, c2, false, false);
    c3 = __builtin_amdgcn_wmma_f32_16x16x32_bf16(false, a, false, b3, (short)0, c3, false, false);
  }

  const v8f cc[4] = {c0, c1, c2, c3};
  #pragma unroll
  for (int t = 0; t < 4; ++t) {
    const int n = nBase + t * 16 + m;
    const float bias = (n < DGT_NODES) ? bp[n] : 0.0f;
    float psum = 0.0f, psq = 0.0f;
    #pragma unroll
    for (int r = 0; r < 8; ++r) {
      float v = cc[t][r] + bias;
      const int row = rowBase + r + 8 * hi;
      predZ[(size_t)row * DGT_NPAD + n] = (__bf16)v;   // bf16 store; stats stay f32
      psum += v;
      psq  += v * v;
    }
    // lanes L and L^16 hold the same column -> pair-reduce, then LDS, then global
    psum += __shfl_xor(psum, 16, 32);
    psq  += __shfl_xor(psq, 16, 32);
    if (hi == 0) { atomicAdd(&sSum[t * 16 + m], psum); atomicAdd(&sSq[t * 16 + m], psq); }
  }
  __syncthreads();
  if (tid < 64) {
    atomicAdd(&colSum[nBase + tid], sSum[tid]);
    atomicAdd(&colSq[nBase + tid],  sSq[tid]);
  }
}

// ---------------- stage 2: fold BN + sigslope into per-node scale/shift ----------------
__global__ __launch_bounds__(256) void dgt_bn_finalize(
    const float* __restrict__ colSum, const float* __restrict__ colSq,
    const float* __restrict__ gamma, const float* __restrict__ beta,
    const float* __restrict__ sigslope,
    float* __restrict__ sArr, float* __restrict__ tArr)
{
  int i = blockIdx.x * 256 + threadIdx.x;
  if (i >= DGT_NPAD) return;
  if (i < DGT_NODES) {
    const float inv = 1.0f / (float)DGT_BATCH;
    float mu  = colSum[i] * inv;
    float var = colSq[i] * inv - mu * mu;
    float rs  = __builtin_amdgcn_rsqf(var + 1e-5f);
    float g   = gamma[i] * rs;
    sArr[i] = g * sigslope[i];
    tArr[i] = (beta[i] - mu * g) * sigslope[i];
  } else {
    sArr[i] = 0.0f;  // padded node -> pred_a == 0
    tArr[i] = 0.0f;
  }
}

// ---------------- stage 3: fused BN-apply + sigmoid + tree-AND + OR WMMA ----------------
// grid = BATCH/16; 256 threads = 8 waves. LDS: predA(32K) | andA(32K); partials alias predA.
__global__ __launch_bounds__(256) void dgt_fused_tail(
    const __bf16* __restrict__ predZ,
    const float* __restrict__ sArr, const float* __restrict__ tArr,
    const float* __restrict__ band, const float* __restrict__ addl,
    const __bf16* __restrict__ WorB, const float* __restrict__ bor,
    float* __restrict__ out)
{
  __shared__ __align__(16) unsigned char smem[65536];
  __bf16* predA = (__bf16*)smem;              // [16][1024] bf16
  __bf16* andA  = (__bf16*)(smem + 32768);    // [16][1024] bf16
  float*  part  = (float*)smem;               // [8][16][16] f32 (reuses predA area)

  const int tid = threadIdx.x;
  const int rowBase = blockIdx.x * 16;

  // --- step 1: BN apply + sigmoid: pred_a = 2*sigmoid(slope*zn) - 1 (8-wide vectorized) ---
  for (int i = tid; i < 16 * DGT_NPAD / 8; i += 256) {
    const int r  = i >> 7;
    const int n0 = (i & 127) * 8;
    v8bf zv = *(const v8bf*)(predZ + (size_t)(rowBase + r) * DGT_NPAD + n0);
    v8bf pa;
    #pragma unroll
    for (int e = 0; e < 8; ++e) {
      float zz = (float)zv[e] * sArr[n0 + e] + tArr[n0 + e];
      float sig = __builtin_amdgcn_rcpf(1.0f + __builtin_amdgcn_exp2f(-1.44269504088896f * zz));
      pa[e] = (__bf16)(2.0f * sig - 1.0f);
    }
    *(v8bf*)(predA + r * DGT_NPAD + n0) = pa;
  }
  __syncthreads();

  // --- step 2: analytic tree AND (10 signed adds per leaf) + bias + ReLU ---
  for (int i = tid; i < 16 * DGT_LEAVES; i += 256) {
    const int r = i >> 10, leaf = i & 1023;
    float acc = band[leaf] + addl[leaf];
    const __bf16* pa = predA + r * DGT_NPAD;
    #pragma unroll
    for (int lvl = 0; lvl < 10; ++lvl) {
      const int node = (1 << lvl) - 1 + (leaf >> (10 - lvl));
      const float v = (float)pa[node];
      acc += ((leaf >> (9 - lvl)) & 1) ? -v : v;
    }
    andA[i] = (__bf16)fmaxf(acc, 0.0f);
  }
  __syncthreads();

  // --- step 3: OR GEMM [16 x 1024] x [1024 x 32] with WMMA from LDS ---
  const int w    = tid >> 5;
  const int lane = tid & 31;
  const int m    = lane & 15;
  const int hi   = lane >> 4;
  const int ntile = w & 1;        // output-column tile (0..1)
  const int kq    = w >> 1;       // K quarter (0..3), each covers 256 leaves
  const int oBase = ntile * 16;

  const __bf16* aRow = andA + m * DGT_NPAD;                       // LDS
  const __bf16* bRow = WorB + (size_t)(oBase + m) * DGT_LEAVES;   // global

  v8f c = {};
  #pragma unroll
  for (int s = 0; s < 8; ++s) {
    const int kk = kq * 256 + s * 32;
    v8bf a0 = *(const v8bf*)(aRow + kk + hi * 16);       // ds_read_b128
    v8bf a1 = *(const v8bf*)(aRow + kk + hi * 16 + 8);
    v16bf a, b;
    b = *(const v16bf*)(bRow + kk + hi * 16);
    #pragma unroll
    for (int i = 0; i < 8; ++i) { a[i] = a0[i]; a[8 + i] = a1[i]; }
    c = __builtin_amdgcn_wmma_f32_16x16x32_bf16(false, a, false, b,
                                                (short)0, c, false, false);
  }

  // partials into LDS (predA region is dead now; andA still live -> untouched)
  #pragma unroll
  for (int r = 0; r < 8; ++r)
    part[w * 256 + (r + 8 * hi) * 16 + m] = c[r];
  __syncthreads();

  // --- reduce 4 K-quarters, add bor, store ---
  for (int e = tid; e < 16 * DGT_OUT; e += 256) {
    const int row = e >> 5, o = e & 31;
    const int nt = o >> 4, col = o & 15;
    float sum = bor[o];
    #pragma unroll
    for (int q = 0; q < 4; ++q)
      sum += part[(q * 2 + nt) * 256 + row * 16 + col];
    out[(size_t)(rowBase + row) * DGT_OUT + o] = sum;
  }
}

// ---------------- host-side launch ----------------
extern "C" void kernel_launch(void* const* d_in, const int* in_sizes, int n_in,
                              void* d_out, int out_size, void* d_ws, size_t ws_size,
                              hipStream_t stream) {
  (void)in_sizes; (void)n_in; (void)out_size; (void)ws_size;
  const float* x        = (const float*)d_in[0];
  const float* sigslope = (const float*)d_in[1];
  // d_in[2] = softslope (unused in ReLU branch)
  const float* addl     = (const float*)d_in[3];
  const float* Wp       = (const float*)d_in[4];
  const float* bp       = (const float*)d_in[5];
  const float* gamma    = (const float*)d_in[6];
  const float* beta     = (const float*)d_in[7];
  // d_in[8] = Wand (fixed tree weight; computed analytically)
  const float* band     = (const float*)d_in[9];
  const float* Wor      = (const float*)d_in[10];
  const float* bor      = (const float*)d_in[11];
  float* out = (float*)d_out;

  char* ws = (char*)d_ws;
  __bf16* Xb     = (__bf16*)(ws + OFF_XB);
  __bf16* Wpb    = (__bf16*)(ws + OFF_WPB);
  __bf16* Worb   = (__bf16*)(ws + OFF_WORB);
  float*  colSum = (float*)(ws + OFF_SUM);
  float*  colSq  = (float*)(ws + OFF_SQ);
  float*  sArr   = (float*)(ws + OFF_S);
  float*  tArr   = (float*)(ws + OFF_T);
  __bf16* predZ  = (__bf16*)(ws + OFF_PZ);

  // prep: bf16 conversions (Wp zero-padded to 1024 rows) + zero BN accumulators
  dgt_cvt_bf16<<<2048, 256, 0, stream>>>(x,   Xb,   DGT_BATCH * DGT_INDIM, DGT_BATCH * DGT_INDIM);
  dgt_cvt_bf16<<<1024, 256, 0, stream>>>(Wp,  Wpb,  DGT_NODES * DGT_INDIM, DGT_NPAD * DGT_INDIM);
  dgt_cvt_bf16<<<128,  256, 0, stream>>>(Wor, Worb, DGT_OUT * DGT_LEAVES,  DGT_OUT * DGT_LEAVES);
  dgt_zero_f32<<<8, 256, 0, stream>>>(colSum, 2048);  // colSum + colSq are contiguous

  // stage 1: pred GEMM + BN statistics (16x64 per wave, 4x A reuse)
  dgt_pred_gemm<<<dim3(DGT_BATCH / 128, DGT_NPAD / 64), 256, 0, stream>>>(
      Xb, Wpb, bp, predZ, colSum, colSq);

  // stage 2: fold BN into scale/shift
  dgt_bn_finalize<<<4, 256, 0, stream>>>(colSum, colSq, gamma, beta, sigslope, sArr, tArr);

  // stage 3: fused sigmoid + tree + OR GEMM
  dgt_fused_tail<<<DGT_BATCH / 16, 256, 0, stream>>>(
      predZ, sArr, tArr, band, addl, Worb, bor, out);
}
